// TokenChoiceMoRLayer_12567074308594
// MI455X (gfx1250) — matrix operand order repaired and checked by reading, not compiled, not run on hardware
//
#include <hip/hip_runtime.h>
#include <hip/hip_bf16.h>
#include <math.h>

// Problem constants (match reference)
#define BB 2
#define TT 4096
#define DD 1024
#define HH 16
#define DHH 64
#define FFD 4096
#define KC 2048
#define LLY 2

typedef __bf16 bh;
typedef __attribute__((ext_vector_type(16))) __bf16 v16bf;
typedef __attribute__((ext_vector_type(8)))  float  v8f;

static __device__ __forceinline__ v8f wmma_bf16(v16bf a, v16bf b, v8f c) {
    return __builtin_amdgcn_wmma_f32_16x16x32_bf16(false, a, false, b, (short)0, c, false, false);
}

union FragU { v16bf v; unsigned u[8]; };

// ---------------------------------------------------------------- async global->LDS (CDNA5)

#if __has_builtin(__builtin_amdgcn_global_load_async_to_lds_b128)
#define USE_ASYNC 1
#else
#define USE_ASYNC 0
#endif

#define GAS1 __attribute__((address_space(1)))
#define LAS3 __attribute__((address_space(3)))
typedef int v4i_g __attribute__((vector_size(16)));   // matches builtin param pointee

// 16B per lane, per-lane global src + per-lane LDS dst (flat->offset via trunc)
static __device__ __forceinline__ void g2l_b128(const bh* gsrc, bh* ldst) {
#if USE_ASYNC
    __builtin_amdgcn_global_load_async_to_lds_b128(
        (GAS1 v4i_g*)(unsigned long long)(size_t)gsrc,
        (LAS3 v4i_g*)(unsigned)(size_t)ldst,
        0, 0);
#else
    *(uint4*)ldst = *(const uint4*)gsrc;
#endif
}

static __device__ __forceinline__ void wait_async() {
#if USE_ASYNC
#if __has_builtin(__builtin_amdgcn_s_wait_asynccnt)
    __builtin_amdgcn_s_wait_asynccnt(0);
#else
    asm volatile("s_wait_asynccnt 0" ::: "memory");
#endif
#endif
}

// ---------------------------------------------------------------- utilities

__global__ void f32_to_bf16_kernel(const float* __restrict__ src, bh* __restrict__ dst, long n) {
    long i = (long)blockIdx.x * blockDim.x + threadIdx.x;
    long stride = (long)gridDim.x * blockDim.x;
    for (; i < n; i += stride) dst[i] = (bh)src[i];
}

// B-weight cast + swizzle into the LDS tile image: dst[((k>>5)*N + n)*32 + (k&31)] = src[k*N + n]
__global__ void wswz_kernel(const float* __restrict__ src, bh* __restrict__ dst,
                            long Kd, long N) {
    long total = Kd * N;
    long i = (long)blockIdx.x * blockDim.x + threadIdx.x;
    long stride = (long)gridDim.x * blockDim.x;
    for (; i < total; i += stride) {
        long k = i / N;
        long n = i - k * N;
        dst[((k >> 5) * N + n) * 32 + (k & 31)] = (bh)src[i];
    }
}

__global__ void copy4_kernel(const float* __restrict__ src, float* __restrict__ dst, long n4) {
    long i = (long)blockIdx.x * blockDim.x + threadIdx.x;
    long stride = (long)gridDim.x * blockDim.x;
    const float4* s = (const float4*)src;
    float4* d = (float4*)dst;
    for (; i < n4; i += stride) d[i] = s[i];
}

// ---------------------------------------------------------------- router

__launch_bounds__(256)
__global__ void router_kernel(const float* __restrict__ x, const float* __restrict__ rw,
                              const float* __restrict__ rb, float* __restrict__ logits) {
    int row = blockIdx.x;          // 0 .. B*T-1
    int tid = threadIdx.x;
    const float* xr = x + (size_t)row * DD;
    float s = 0.f;
    for (int d = tid; d < DD; d += 256) s += xr[d] * rw[d];
    __shared__ float red[256];
    red[tid] = s; __syncthreads();
    for (int off = 128; off > 0; off >>= 1) {
        if (tid < off) red[tid] += red[tid + off];
        __syncthreads();
    }
    if (tid == 0) logits[row] = red[0] + rb[0];
}

// ---------------------------------------------------------------- top-k (bitonic, per batch)

__launch_bounds__(1024)
__global__ void topk_kernel(const float* __restrict__ logits, int* __restrict__ selidx,
                            float* __restrict__ gates) {
    int b = blockIdx.x;
    int tid = threadIdx.x;
    __shared__ float sv[TT];
    __shared__ int   si[TT];
    for (int i = tid; i < TT; i += 1024) { sv[i] = logits[b * TT + i]; si[i] = i; }
    __syncthreads();
    for (int k = 2; k <= TT; k <<= 1) {
        for (int j = k >> 1; j > 0; j >>= 1) {
            for (int i = tid; i < TT; i += 1024) {
                int ixj = i ^ j;
                if (ixj > i) {
                    float a = sv[i], c = sv[ixj];
                    int ia = si[i], ic = si[ixj];
                    bool up = ((i & k) == 0);                       // descending blocks
                    bool greater = (a > c) || (a == c && ia < ic);  // deterministic tiebreak
                    bool sw = up ? !greater : greater;
                    if (sw) { sv[i] = c; sv[ixj] = a; si[i] = ic; si[ixj] = ia; }
                }
            }
            __syncthreads();
        }
    }
    for (int i = tid; i < KC; i += 1024) {
        selidx[b * KC + i] = si[i];
        gates[b * KC + i]  = 1.f / (1.f + __expf(-sv[i]));
    }
}

// ---------------------------------------------------------------- gather / scatter

__launch_bounds__(256)
__global__ void gather_kernel(const float* __restrict__ x, const int* __restrict__ sel,
                              float* __restrict__ h) {
    int row = blockIdx.x;                 // 0 .. B*KC-1
    int b = row / KC;
    int t = sel[row];
    const float4* src = (const float4*)(x + ((size_t)b * TT + t) * DD);
    float4* dst = (float4*)(h + (size_t)row * DD);
    for (int i = threadIdx.x; i < DD / 4; i += 256) dst[i] = src[i];
}

__launch_bounds__(256)
__global__ void scatter_kernel(const float* __restrict__ h, const int* __restrict__ sel,
                               const float* __restrict__ gate, float* __restrict__ out) {
    int row = blockIdx.x;
    int b = row / KC;
    int t = sel[row];
    float g = gate[row];
    float* dst = out + ((size_t)b * TT + t) * DD;
    const float* src = h + (size_t)row * DD;
    for (int d = threadIdx.x; d < DD; d += 256) dst[d] += src[d] * g;
}

// ---------------------------------------------------------------- layernorm (f32 in, bf16 out)

__launch_bounds__(256)
__global__ void ln_kernel(const float* __restrict__ h, const float* __restrict__ w,
                          const float* __restrict__ bias, bh* __restrict__ y) {
    int row = blockIdx.x;
    int tid = threadIdx.x;
    const float* hr = h + (size_t)row * DD;
    float v[4];
    float s = 0.f;
#pragma unroll
    for (int i = 0; i < 4; ++i) { v[i] = hr[tid + i * 256]; s += v[i]; }
    __shared__ float red[256];
    red[tid] = s; __syncthreads();
    for (int off = 128; off > 0; off >>= 1) { if (tid < off) red[tid] += red[tid + off]; __syncthreads(); }
    float mu = red[0] * (1.f / DD);
    __syncthreads();
    float q = 0.f;
#pragma unroll
    for (int i = 0; i < 4; ++i) { float d = v[i] - mu; q += d * d; }
    red[tid] = q; __syncthreads();
    for (int off = 128; off > 0; off >>= 1) { if (tid < off) red[tid] += red[tid + off]; __syncthreads(); }
    float rstd = rsqrtf(red[0] * (1.f / DD) + 1e-5f);
    bh* yr = y + (size_t)row * DD;
#pragma unroll
    for (int i = 0; i < 4; ++i) {
        int d = tid + i * 256;
        yr[d] = (bh)((v[i] - mu) * rstd * w[d] + bias[d]);
    }
}

// ---------------------------------------------------------------- WMMA GEMM  C = A(MxK) * Bswz(KxN)
// B must be pre-swizzled: Bp[((k>>5)*N + n)*32 + (k&31)]
// MODE 0: outH = bf16(acc)            (QKV)
// MODE 1: outF = resid + acc  (f32)   (WO / W2 residual)
// MODE 2: outH = bf16(gelu(acc))      (W1)

#define BM 128
#define BN 128
#define BK 32
#define LDA 40
#define LDB 40

template <int MODE>
__launch_bounds__(256)
__global__ void gemm_bf16_kernel(const bh* __restrict__ A, const bh* __restrict__ Bp,
                                 int M, int N, int Kd,
                                 const float* __restrict__ resid,
                                 float* __restrict__ outF, bh* __restrict__ outH) {
    __shared__ bh sA[2][BM * LDA];   // [row][k]
    __shared__ bh sB[2][BN * LDB];   // [n][k]

    int tiles_n = N / BN;
    int bm = blockIdx.x / tiles_n;
    int bn = blockIdx.x % tiles_n;

    int tid  = threadIdx.x;
    int lane = tid & 31;
    int wave = tid >> 5;
    int wm = wave >> 1;             // 0..3  (32-row strip)
    int wn = wave & 1;              // 0..1  (64-col strip)
    int g  = lane >> 4;
    int mr = lane & 15;

    v8f acc[2][4];
#pragma unroll
    for (int i = 0; i < 2; ++i)
#pragma unroll
        for (int j = 0; j < 4; ++j) acc[i][j] = (v8f)0.f;

    auto stage = [&](int k0, int buf) {
        // A tile: 128x32 = 512 x (8 bf16) lane-transfers
#pragma unroll
        for (int r = 0; r < 2; ++r) {
            int lin = tid * 2 + r;            // 0..511
            int row = lin >> 2;               // 4 vec8 per row
            int kc  = (lin & 3) << 3;
            g2l_b128(A + (size_t)(bm * BM + row) * Kd + k0 + kc,
                     &sA[buf][row * LDA + kc]);
        }
        // B tile: pre-swizzled [n][k32] -> contiguous lane transfers
#pragma unroll
        for (int r = 0; r < 2; ++r) {
            int lin = tid * 2 + r;            // 0..511
            int n  = lin >> 2;
            int kc = (lin & 3) << 3;
            g2l_b128(Bp + ((size_t)(k0 >> 5) * N + bn * BN + n) * 32 + kc,
                     &sB[buf][n * LDB + kc]);
        }
    };

    const int nk = Kd / BK;
    stage(0, 0);
    wait_async();
    __syncthreads();

    for (int t = 0; t < nk; ++t) {
        int cur = t & 1;
        if (t + 1 < nk) stage((t + 1) * BK, cur ^ 1);   // prefetch next tile

        v16bf af[2], bfr[4];
#pragma unroll
        for (int i = 0; i < 2; ++i) {
            FragU u;
            int row = wm * 32 + i * 16 + mr;
#pragma unroll
            for (int p = 0; p < 8; ++p) {
                int kb = ((p >= 4) ? 16 : 0) + g * 8 + (p & 3) * 2;
                u.u[p] = *(const unsigned*)&sA[cur][row * LDA + kb];
            }
            af[i] = u.v;
        }
#pragma unroll
        for (int j = 0; j < 4; ++j) {
            FragU u;
            int col = wn * 64 + j * 16 + mr;
#pragma unroll
            for (int p = 0; p < 8; ++p) {
                int kb = ((p >= 4) ? 16 : 0) + g * 8 + (p & 3) * 2;
                u.u[p] = *(const unsigned*)&sB[cur][col * LDB + kb];
            }
            bfr[j] = u.v;
        }
#pragma unroll
        for (int i = 0; i < 2; ++i)
#pragma unroll
            for (int j = 0; j < 4; ++j)
                acc[i][j] = wmma_bf16(af[i], bfr[j], acc[i][j]);

        wait_async();
        __syncthreads();
    }

    // epilogue
#pragma unroll
    for (int i = 0; i < 2; ++i)
#pragma unroll
        for (int j = 0; j < 4; ++j)
#pragma unroll
            for (int p = 0; p < 8; ++p) {
                int row = bm * BM + wm * 32 + i * 16 + g * 8 + p;
                int col = bn * BN + wn * 64 + j * 16 + mr;
                size_t idx = (size_t)row * N + col;
                float val = acc[i][j][p];
                if (MODE == 0) {
                    outH[idx] = (bh)val;
                } else if (MODE == 1) {
                    outF[idx] = resid[idx] + val;
                } else {
                    float tg = tanhf(0.7978845608028654f * (val + 0.044715f * val * val * val));
                    outH[idx] = (bh)(0.5f * val * (1.f + tg));
                }
            }
}

// ---------------------------------------------------------------- WMMA flash attention
// qkv: [B*KC, 3*DD] bf16; per (b, head, q-tile of 128): o: [B*KC, DD] bf16

#define QV_STRIDE (3 * DD)
#define LSK 72      // K tile [key][dh]
#define LSV 136     // V tile [dh][key] (transposed)
#define LSP 136     // P strip [row][key]

__launch_bounds__(256)
__global__ void attn_kernel(const bh* __restrict__ qkv, bh* __restrict__ o) {
    const int nqt = KC / 128;             // 16
    int b   = blockIdx.x / (HH * nqt);
    int rem = blockIdx.x % (HH * nqt);
    int hd  = rem / nqt;
    int qt  = rem % nqt;

    __shared__ bh sK[128 * LSK];
    __shared__ bh sV[DHH * LSV];
    __shared__ bh sP[8 * 16 * LSP];

    int tid  = threadIdx.x;
    int lane = tid & 31;
    int wave = tid >> 5;
    int g  = lane >> 4;
    int mr = lane & 15;

    const size_t rowbase = (size_t)b * KC;
    const int qrow0 = qt * 128 + wave * 16;

    // Q fragments (loop-invariant), k-dim = DHH split into 2 halves of 32
    v16bf aq[2];
#pragma unroll
    for (int i = 0; i < 2; ++i) {
        FragU t;
        size_t qoff = (rowbase + qrow0 + mr) * QV_STRIDE + hd * DHH;
#pragma unroll
        for (int p = 0; p < 8; ++p) {
            int kb = i * 32 + ((p >= 4) ? 16 : 0) + g * 8 + (p & 3) * 2;
            t.u[p] = *(const unsigned*)&qkv[qoff + kb];
        }
        aq[i] = t.v;
    }

    v8f oacc[4];
#pragma unroll
    for (int j = 0; j < 4; ++j) oacc[j] = (v8f)0.f;
    float m_i[8], l_i[8];
#pragma unroll
    for (int p = 0; p < 8; ++p) { m_i[p] = -1e30f; l_i[p] = 0.f; }

    for (int kt = 0; kt < KC / 128; ++kt) {
        // stage K tile [key][dh] (async) and V tile transposed [dh][key]
#pragma unroll
        for (int r = 0; r < 4; ++r) {
            int v = tid + r * 256;            // 0..1023 vec8 indices
            int key = v >> 3;
            int dh  = (v & 7) << 3;
            size_t roff = (rowbase + kt * 128 + key) * QV_STRIDE + hd * DHH + dh;
            g2l_b128(qkv + roff + DD, &sK[key * LSK + dh]);          // K async
            uint4 dv = *(const uint4*)&qkv[roff + 2 * DD];           // V
            const bh* e = (const bh*)&dv;
#pragma unroll
            for (int q = 0; q < 8; ++q) sV[(dh + q) * LSV + key] = e[q];
        }
        wait_async();
        __syncthreads();

        // S = Q * K^T  (16 x 128 per wave), scaled
        v8f s[8];
#pragma unroll
        for (int nf = 0; nf < 8; ++nf) {
            s[nf] = (v8f)0.f;
#pragma unroll
            for (int ks = 0; ks < 2; ++ks) {
                FragU t;
                int col = nf * 16 + mr;       // key index = B-matrix column
#pragma unroll
                for (int p = 0; p < 8; ++p) {
                    int kb = ks * 32 + ((p >= 4) ? 16 : 0) + g * 8 + (p & 3) * 2;
                    t.u[p] = *(const unsigned*)&sK[col * LSK + kb];
                }
                s[nf] = wmma_bf16(aq[ks], t.v, s[nf]);
            }
        }
#pragma unroll
        for (int nf = 0; nf < 8; ++nf)
#pragma unroll
            for (int p = 0; p < 8; ++p) s[nf][p] *= 0.125f;   // 1/sqrt(64)

        // online softmax per row (rows m = g*8+p, reduce across 16 column-lanes)
#pragma unroll
        for (int p = 0; p < 8; ++p) {
            float mx = -1e30f;
#pragma unroll
            for (int nf = 0; nf < 8; ++nf) mx = fmaxf(mx, s[nf][p]);
            mx = fmaxf(mx, __shfl_xor(mx, 1, 32));
            mx = fmaxf(mx, __shfl_xor(mx, 2, 32));
            mx = fmaxf(mx, __shfl_xor(mx, 4, 32));
            mx = fmaxf(mx, __shfl_xor(mx, 8, 32));
            float mnew = fmaxf(m_i[p], mx);
            float alpha = __expf(m_i[p] - mnew);
            float rs = 0.f;
#pragma unroll
            for (int nf = 0; nf < 8; ++nf) {
                float e = __expf(s[nf][p] - mnew);
                s[nf][p] = e;
                rs += e;
            }
            rs += __shfl_xor(rs, 1, 32);
            rs += __shfl_xor(rs, 2, 32);
            rs += __shfl_xor(rs, 4, 32);
            rs += __shfl_xor(rs, 8, 32);
            l_i[p] = l_i[p] * alpha + rs;
            m_i[p] = mnew;
#pragma unroll
            for (int j = 0; j < 4; ++j) oacc[j][p] *= alpha;
        }

        // restage P (16x128 bf16) into wave-private LDS, reload as A fragments
        bh* pw = &sP[wave * 16 * LSP];
#pragma unroll
        for (int nf = 0; nf < 8; ++nf)
#pragma unroll
            for (int p = 0; p < 8; ++p)
                pw[(g * 8 + p) * LSP + nf * 16 + mr] = (bh)s[nf][p];
        __syncthreads();

        // O += P * V   (k-dim = 128 keys)
#pragma unroll
        for (int ks = 0; ks < 4; ++ks) {
            FragU ta;
#pragma unroll
            for (int p = 0; p < 8; ++p) {
                int kb = ks * 32 + ((p >= 4) ? 16 : 0) + g * 8 + (p & 3) * 2;
                ta.u[p] = *(const unsigned*)&pw[mr * LSP + kb];
            }
#pragma unroll
            for (int j = 0; j < 4; ++j) {
                FragU tb;
                int col = j * 16 + mr;        // dh column
#pragma unroll
                for (int p = 0; p < 8; ++p) {
                    int kb = ks * 32 + ((p >= 4) ? 16 : 0) + g * 8 + (p & 3) * 2;
                    tb.u[p] = *(const unsigned*)&sV[col * LSV + kb];
                }
                oacc[j] = wmma_bf16(ta.v, tb.v, oacc[j]);
            }
        }
        __syncthreads();
    }

    // finalize
#pragma unroll
    for (int j = 0; j < 4; ++j)
#pragma unroll
        for (int p = 0; p < 8; ++p) {
            size_t row = rowbase + qrow0 + g * 8 + p;
            int col = hd * DHH + j * 16 + mr;
            o[row * DD + col] = (bh)(oacc[j][p] / l_i[p]);
        }
}

// ---------------------------------------------------------------- host orchestration

extern "C" void kernel_launch(void* const* d_in, const int* in_sizes, int n_in,
                              void* d_out, int out_size, void* d_ws, size_t ws_size,
                              hipStream_t stream) {
    (void)in_sizes; (void)n_in; (void)out_size; (void)ws_size;
    const float* x        = (const float*)d_in[0];
    const float* router_w = (const float*)d_in[1];
    const float* router_b = (const float*)d_in[2];
    const float* ln1_w    = (const float*)d_in[3];
    const float* ln1_b    = (const float*)d_in[4];
    const float* wqkv     = (const float*)d_in[5];
    const float* wo       = (const float*)d_in[6];
    const float* ln2_w    = (const float*)d_in[7];
    const float* ln2_b    = (const float*)d_in[8];
    const float* w1       = (const float*)d_in[9];
    const float* w2       = (const float*)d_in[10];
    float* out = (float*)d_out;

    char* ws = (char*)d_ws;
    size_t off = 0;
    auto carve = [&](size_t bytes) -> void* {
        void* p = ws + off;
        off += (bytes + 255) & ~(size_t)255;
        return p;
    };
    float* logits = (float*)carve((size_t)BB * TT * 4);
    int*   selidx = (int*)  carve((size_t)BB * KC * 4);
    float* gates  = (float*)carve((size_t)BB * KC * 4);
    float* h      = (float*)carve((size_t)BB * KC * DD * 4);
    bh* y    = (bh*)carve((size_t)BB * KC * DD * 2);
    bh* qkvb = (bh*)carve((size_t)BB * KC * 3 * DD * 2);
    bh* ob   = (bh*)carve((size_t)BB * KC * DD * 2);
    bh* ffb  = (bh*)carve((size_t)BB * KC * FFD * 2);
    bh* wqkv_h = (bh*)carve((size_t)LLY * DD * 3 * DD * 2);
    bh* wo_h   = (bh*)carve((size_t)LLY * DD * DD * 2);
    bh* w1_h   = (bh*)carve((size_t)LLY * DD * FFD * 2);
    bh* w2_h   = (bh*)carve((size_t)LLY * FFD * DD * 2);

    const int M = BB * KC;   // 4096 active token rows

    // weight casts + swizzle into LDS tile image (per layer slice)
    for (int l = 0; l < LLY; ++l) {
        wswz_kernel<<<1024, 256, 0, stream>>>(wqkv + (size_t)l * DD * 3 * DD,
                                              wqkv_h + (size_t)l * DD * 3 * DD, DD, 3 * DD);
        wswz_kernel<<<512, 256, 0, stream>>>(wo + (size_t)l * DD * DD,
                                             wo_h + (size_t)l * DD * DD, DD, DD);
        wswz_kernel<<<1024, 256, 0, stream>>>(w1 + (size_t)l * DD * FFD,
                                              w1_h + (size_t)l * DD * FFD, DD, FFD);
        wswz_kernel<<<1024, 256, 0, stream>>>(w2 + (size_t)l * FFD * DD,
                                              w2_h + (size_t)l * FFD * DD, FFD, DD);
    }

    // router + top-k + gather; out starts as copy of x
    router_kernel<<<BB * TT, 256, 0, stream>>>(x, router_w, router_b, logits);
    topk_kernel<<<BB, 1024, 0, stream>>>(logits, selidx, gates);
    gather_kernel<<<BB * KC, 256, 0, stream>>>(x, selidx, h);
    copy4_kernel<<<4096, 256, 0, stream>>>(x, out, (long)BB * TT * DD / 4);

    for (int l = 0; l < LLY; ++l) {
        const bh* Wqkv = wqkv_h + (size_t)l * DD * 3 * DD;
        const bh* Wo   = wo_h   + (size_t)l * DD * DD;
        const bh* W1   = w1_h   + (size_t)l * DD * FFD;
        const bh* W2   = w2_h   + (size_t)l * FFD * DD;

        ln_kernel<<<M, 256, 0, stream>>>(h, ln1_w + l * DD, ln1_b + l * DD, y);
        gemm_bf16_kernel<0><<<(M / BM) * (3 * DD / BN), 256, 0, stream>>>(
            y, Wqkv, M, 3 * DD, DD, nullptr, nullptr, qkvb);
        attn_kernel<<<BB * HH * (KC / 128), 256, 0, stream>>>(qkvb, ob);
        gemm_bf16_kernel<1><<<(M / BM) * (DD / BN), 256, 0, stream>>>(
            ob, Wo, M, DD, DD, h, h, nullptr);
        ln_kernel<<<M, 256, 0, stream>>>(h, ln2_w + l * DD, ln2_b + l * DD, y);
        gemm_bf16_kernel<2><<<(M / BM) * (FFD / BN), 256, 0, stream>>>(
            y, W1, M, FFD, DD, nullptr, nullptr, ffb);
        gemm_bf16_kernel<1><<<(M / BM) * (DD / BN), 256, 0, stream>>>(
            ffb, W2, M, DD, FFD, h, h, nullptr);
    }

    scatter_kernel<<<BB * KC, 256, 0, stream>>>(h, selidx, gates, out);
}
